// EdgePredictor_Task2_80152679678507
// MI455X (gfx1250) — matrix operand order repaired and checked by reading, not compiled, not run on hardware
//
#include <hip/hip_runtime.h>
#include <hip/hip_bf16.h>
#include <math.h>

#define DDIM 128
#define BN_EPS 1e-5f

typedef __attribute__((ext_vector_type(16))) __bf16 v16bf;
typedef __attribute__((ext_vector_type(8)))  float  v8f;

// ---------------------------------------------------------------------------
// pack_a: convert H (fp32, row-major [nRows x 128]) into bf16 in the exact
// WMMA A-operand layout. One thread per (rowTile, kChunk, lane) writes 16
// contiguous bf16. Thread id t == ((rowTile*4 + kChunk)*32 + lane).
// A 16x32 layout (ISA 7.12.2): lane = M (mod 16), half = lane>>4,
//   VGPR v holds K = (v>=4?16:0) + 8*half + 2*(v&3)  (+ kChunk*32)
// ---------------------------------------------------------------------------
__global__ void __launch_bounds__(256)
pack_a(const float* __restrict__ H, __bf16* __restrict__ aP, int nRows)
{
    const int t = blockIdx.x * blockDim.x + threadIdx.x;
    const int lane    = t & 31;
    const int kChunk  = (t >> 5) & 3;
    const int rowTile = t >> 7;
    if (rowTile * 16 >= nRows) return;

    const int m    = lane & 15;
    const int half = lane >> 4;
    const float* src = H + (size_t)(rowTile * 16 + m) * DDIM + kChunk * 32;
    __bf16* dst = aP + (size_t)t * 16;

    #pragma unroll
    for (int v = 0; v < 8; ++v) {
        const int ka = ((v & 4) ? 16 : 0) + half * 8 + (v & 3) * 2;
        dst[2 * v]     = (__bf16)src[ka];
        dst[2 * v + 1] = (__bf16)src[ka + 1];
    }
}

// ---------------------------------------------------------------------------
// pack_b: convert both weight matrices (fp32 [128x128] row-major) into bf16
// WMMA B-operand layout. t == ((mat*8 + colTile)*4 + kChunk)*32 + lane,
// 2048 threads total. B 32x16 layout: lane = N (mod 16), half = lane>>4,
//   VGPR v holds K = (v>=4?16:0) + 8*half + 2*(v&3) (+ kChunk*32), column N.
// ---------------------------------------------------------------------------
__global__ void __launch_bounds__(256)
pack_b(const float* __restrict__ W, const float* __restrict__ Wself,
       __bf16* __restrict__ bP)
{
    const int t = blockIdx.x * blockDim.x + threadIdx.x;
    if (t >= 2048) return;

    const int lane    = t & 31;
    const int kChunk  = (t >> 5) & 3;
    const int colTile = (t >> 7) & 7;
    const float* M = (t >> 10) ? Wself : W;

    const int half = lane >> 4;
    const int col  = colTile * 16 + (lane & 15);
    __bf16* dst = bP + (size_t)t * 16;

    #pragma unroll
    for (int v = 0; v < 8; ++v) {
        const int kb = kChunk * 32 + ((v & 4) ? 16 : 0) + half * 8 + (v & 3) * 2;
        dst[2 * v]     = (__bf16)M[kb * DDIM + col];
        dst[2 * v + 1] = (__bf16)M[(kb + 1) * DDIM + col];
    }
}

// ---------------------------------------------------------------------------
// Fused dual GEMM from packed operands:
//   support = H @ W ; outb = H @ Wself + bias
// One wave per 16x16 output tile. Inner loop: 3 contiguous 32B loads + 2 WMMA.
// ---------------------------------------------------------------------------
__global__ void __launch_bounds__(256)
gemm_dual_wmma(const __bf16* __restrict__ aP,
               const __bf16* __restrict__ bP,
               const float* __restrict__ bias,
               float* __restrict__ support,
               float* __restrict__ outb,
               int nRows)
{
    const int wavesPerBlock = blockDim.x >> 5;
    const int waveId = blockIdx.x * wavesPerBlock + (threadIdx.x >> 5);
    const int lane   = threadIdx.x & 31;
    const int numColTiles = DDIM / 16;                 // 8
    const int rowTile = waveId / numColTiles;
    const int colTile = waveId % numColTiles;
    if (rowTile * 16 >= nRows) return;

    // packed bases: groups of 512 bf16 (32 lanes x 16) per (tile, kChunk)
    const __bf16* aBase  = aP + (((size_t)rowTile * 4) * 32 + lane) * 16;
    const __bf16* bSBase = bP + (((size_t)(0 * 8 + colTile) * 4) * 32 + lane) * 16;
    const __bf16* bFBase = bP + (((size_t)(1 * 8 + colTile) * 4) * 32 + lane) * 16;

    v8f accS = {};                                     // h @ W
    v8f accF = {};                                     // h @ Wself

    #pragma unroll
    for (int kcI = 0; kcI < 4; ++kcI) {
        const v16bf a  = *(const v16bf*)(aBase  + (size_t)kcI * 512);
        const v16bf bS = *(const v16bf*)(bSBase + (size_t)kcI * 512);
        const v16bf bF = *(const v16bf*)(bFBase + (size_t)kcI * 512);
        accS = __builtin_amdgcn_wmma_f32_16x16x32_bf16(
                   false, a, false, bS, (short)0, accS, false, false);
        accF = __builtin_amdgcn_wmma_f32_16x16x32_bf16(
                   false, a, false, bF, (short)0, accF, false, false);
    }

    // C/D layout: lane carries N = lane&15, VGPR r carries M = r + 8*half
    const int halfSel = lane >> 4;
    const int cn      = colTile * 16 + (lane & 15);
    const int mBase   = rowTile * 16 + halfSel * 8;
    const float bval  = bias[cn];
    #pragma unroll
    for (int r = 0; r < 8; ++r) {
        const int cm = mBase + r;
        support[cm * DDIM + cn] = accS[r];
        outb[cm * DDIM + cn]    = accF[r] + bval;      // self-term + bias seed
    }
}

// ---------------------------------------------------------------------------
// SpMM scatter-add: outb[row] += val * support[col]
// 32 lanes per edge, float4 per lane, global f32 atomics (L2-resident).
// ---------------------------------------------------------------------------
__global__ void __launch_bounds__(256)
spmm_scatter(const int* __restrict__ adj_row,
             const int* __restrict__ adj_col,
             const float* __restrict__ adj_val,
             const float* __restrict__ support,
             float* __restrict__ outb,
             int nnz)
{
    const long long t = (long long)blockIdx.x * blockDim.x + threadIdx.x;
    const int e    = (int)(t >> 5);
    const int lane = (int)(t & 31);
    if (e >= nnz) return;

    const int   r = adj_row[e];
    const int   c = adj_col[e];
    const float v = adj_val[e];

    const float4 s = ((const float4*)(support + (size_t)c * DDIM))[lane];
    float* dst = outb + (size_t)r * DDIM + lane * 4;
    atomicAdd(dst + 0, v * s.x);
    atomicAdd(dst + 1, v * s.y);
    atomicAdd(dst + 2, v * s.z);
    atomicAdd(dst + 3, v * s.w);
}

// ---------------------------------------------------------------------------
// BatchNorm column statistics: sums[0:D] = sum, sums[D:2D] = sum of squares.
// blockDim.x == DDIM, each thread owns one column, rows strided by gridDim.
// ---------------------------------------------------------------------------
__global__ void __launch_bounds__(DDIM)
bn_stats(const float* __restrict__ outb, float* __restrict__ sums, int n)
{
    const int col = threadIdx.x;
    float s = 0.f, s2 = 0.f;
    for (int r = blockIdx.x; r < n; r += gridDim.x) {
        const float v = outb[(size_t)r * DDIM + col];
        s  += v;
        s2 += v * v;
    }
    atomicAdd(&sums[col], s);
    atomicAdd(&sums[DDIM + col], s2);
}

// ---------------------------------------------------------------------------
// Fused BN-normalize + ReLU (+ optional residual add of x for the last layer)
// ---------------------------------------------------------------------------
__global__ void __launch_bounds__(256)
bn_apply_relu(const float* __restrict__ outb,
              const float* __restrict__ sums,
              const float* __restrict__ gamma,
              const float* __restrict__ beta,
              const float* __restrict__ x,
              float* __restrict__ h,
              int n, int addResidual)
{
    const long long idx = (long long)blockIdx.x * blockDim.x + threadIdx.x;
    if (idx >= (long long)n * DDIM) return;
    const int c = (int)(idx & (DDIM - 1));

    const float invN = 1.0f / (float)n;
    const float mu   = sums[c] * invN;
    const float var  = sums[DDIM + c] * invN - mu * mu;   // biased variance
    float v = gamma[c] * (outb[idx] - mu) * rsqrtf(var + BN_EPS) + beta[c];
    v = fmaxf(v, 0.0f);
    if (addResidual) v += x[idx];
    h[idx] = v;
}

// ---------------------------------------------------------------------------
// Edge predictor: sigmoid(emb[i].Wc[0:128] + emb[j].Wc[128:256] + bc)
// One wave32 per edge: float4 per lane covers both 128-wide dots, then a
// 5-step shuffle reduction.
// ---------------------------------------------------------------------------
__global__ void __launch_bounds__(256)
edge_predict(const float* __restrict__ emb,
             const int* __restrict__ edge_index,
             const float* __restrict__ Wc,
             const float* __restrict__ bc,
             float* __restrict__ outp,
             int nE)
{
    const long long t = (long long)blockIdx.x * blockDim.x + threadIdx.x;
    const int e    = (int)(t >> 5);
    const int lane = (int)(t & 31);
    if (e >= nE) return;

    const int i = edge_index[e];
    const int j = edge_index[nE + e];

    const float4 a  = ((const float4*)(emb + (size_t)i * DDIM))[lane];
    const float4 b  = ((const float4*)(emb + (size_t)j * DDIM))[lane];
    const float4 w1 = ((const float4*)Wc)[lane];
    const float4 w2 = ((const float4*)(Wc + DDIM))[lane];

    float acc = a.x * w1.x + a.y * w1.y + a.z * w1.z + a.w * w1.w
              + b.x * w2.x + b.y * w2.y + b.z * w2.z + b.w * w2.w;

    #pragma unroll
    for (int off = 16; off > 0; off >>= 1)
        acc += __shfl_down(acc, off, 32);

    if (lane == 0) {
        const float z = acc + bc[0];
        outp[e] = 1.0f / (1.0f + __expf(-z));
    }
}

// ---------------------------------------------------------------------------
// Host-side orchestration (graph-capture safe: only kernels + async memset)
// ---------------------------------------------------------------------------
extern "C" void kernel_launch(void* const* d_in, const int* in_sizes, int n_in,
                              void* d_out, int out_size, void* d_ws, size_t ws_size,
                              hipStream_t stream)
{
    const float* x          = (const float*)d_in[0];
    const int*   adj_row    = (const int*)  d_in[1];
    const int*   adj_col    = (const int*)  d_in[2];
    const float* adj_val    = (const float*)d_in[3];
    const int*   edge_index = (const int*)  d_in[4];
    const float* W          = (const float*)d_in[5];
    const float* W_self     = (const float*)d_in[6];
    const float* b          = (const float*)d_in[7];
    const float* gamma      = (const float*)d_in[8];
    const float* beta       = (const float*)d_in[9];
    const float* Wc         = (const float*)d_in[10];
    const float* bc         = (const float*)d_in[11];

    const int N      = in_sizes[0] / DDIM;          // 50000
    const int E_ADJ  = in_sizes[1];                 // 800000
    const int E_PRED = in_sizes[4] / 2;             // 500000
    const int L      = in_sizes[5] / (DDIM * DDIM); // 2

    // Workspace carve-up: aPacked (bf16) | bPacked (bf16) | support | out | h | sums
    __bf16* aPacked = (__bf16*)d_ws;
    __bf16* bPacked = aPacked + (size_t)N * DDIM;
    float*  support = (float*)(bPacked + (size_t)2 * DDIM * DDIM);
    float*  outb    = support + (size_t)N * DDIM;
    float*  hbuf    = outb    + (size_t)N * DDIM;
    float*  sums    = hbuf    + (size_t)N * DDIM;

    const int tiles      = (N / 16) * (DDIM / 16);          // 25000 waves
    const int gemmBlocks = (tiles + 7) / 8;                 // 8 waves/block
    const int packABlks  = ((N / 16) * 128 + 255) / 256;    // 1 thread per 16 bf16
    const long long spmmThreads = (long long)E_ADJ * 32;
    const int spmmBlocks = (int)((spmmThreads + 255) / 256);
    const long long elems = (long long)N * DDIM;
    const int applyBlocks = (int)((elems + 255) / 256);
    const long long edgeThreads = (long long)E_PRED * 32;
    const int edgeBlocks = (int)((edgeThreads + 255) / 256);

    const float* h = x;
    for (int l = 0; l < L; ++l) {
        hipMemsetAsync(sums, 0, 2 * DDIM * sizeof(float), stream);

        pack_a<<<packABlks, 256, 0, stream>>>(h, aPacked, N);
        pack_b<<<8, 256, 0, stream>>>(
            W + (size_t)l * DDIM * DDIM, W_self + (size_t)l * DDIM * DDIM, bPacked);

        gemm_dual_wmma<<<gemmBlocks, 256, 0, stream>>>(
            aPacked, bPacked, b + (size_t)l * DDIM, support, outb, N);

        spmm_scatter<<<spmmBlocks, 256, 0, stream>>>(
            adj_row, adj_col, adj_val, support, outb, E_ADJ);

        bn_stats<<<256, DDIM, 0, stream>>>(outb, sums, N);

        bn_apply_relu<<<applyBlocks, 256, 0, stream>>>(
            outb, sums, gamma + (size_t)l * DDIM, beta + (size_t)l * DDIM,
            x, hbuf, N, (l == L - 1) ? 1 : 0);

        h = hbuf;
    }

    edge_predict<<<edgeBlocks, 256, 0, stream>>>(
        hbuf, edge_index, Wc, bc, (float*)d_out, E_PRED);
}
